// SentenceEncodingLayer_69638599737531
// MI455X (gfx1250) — matrix operand order repaired and checked by reading, not compiled, not run on hardware
//
#include <hip/hip_runtime.h>
#include <math.h>

typedef __attribute__((ext_vector_type(16))) _Float16 v16h;
typedef __attribute__((ext_vector_type(8)))  _Float16 v8h;
typedef __attribute__((ext_vector_type(8)))  float    v8f;

#define TT    256
#define BB    64
#define II    512
#define HH    512
#define FOURH 2048

__device__ __forceinline__ float sigmoidf_(float x) { return 1.0f / (1.0f + __expf(-x)); }

// A fragment: lane holds row m; halves = K0+hi*8+[0..7] and K0+16+hi*8+[0..7]
__device__ __forceinline__ v16h ldA(const _Float16* p) {
  v8h lo = *reinterpret_cast<const v8h*>(p);
  v8h hi = *reinterpret_cast<const v8h*>(p + 16);
  v16h a;
#pragma unroll
  for (int e = 0; e < 8; ++e) { a[e] = lo[e]; a[e + 8] = hi[e]; }
  return a;
}

// B fragment: lane holds column g; halves = K0+hi*8+[0..7] and K0+16+hi*8+[0..7]
__device__ __forceinline__ v16h ldB(const _Float16* p) {
  v8h lo = *reinterpret_cast<const v8h*>(p);
  v8h hi = *reinterpret_cast<const v8h*>(p + 16);
  v16h b;
#pragma unroll
  for (int e = 0; e < 8; ++e) { b[e] = lo[e]; b[e + 8] = hi[e]; }
  return b;
}

#define WMMA_F16(A, B, C) \
  __builtin_amdgcn_wmma_f32_16x16x32_f16(false, (A), false, (B), (short)0, (C), false, false)

// ---------------- prep kernels ----------------

__global__ void cvt_f16_kernel(const float* __restrict__ s, _Float16* __restrict__ d, size_t n) {
  size_t i = (size_t)blockIdx.x * blockDim.x + threadIdx.x;
  size_t st = (size_t)gridDim.x * blockDim.x;
  for (; i < n; i += st) d[i] = (_Float16)s[i];
}

// dst[row][0..511] = whh[row][k] ; dst[row][512..1023] = wch[row][k]   (row = dir*4H + g)
__global__ void build_wrec_kernel(const float* __restrict__ whh, const float* __restrict__ wch,
                                  _Float16* __restrict__ d, size_t n) {
  size_t i = (size_t)blockIdx.x * blockDim.x + threadIdx.x;
  size_t st = (size_t)gridDim.x * blockDim.x;
  for (; i < n; i += st) {
    size_t row = i >> 10;
    int k = (int)(i & 1023);
    float v = (k < HH) ? whh[row * HH + k] : wch[row * HH + (k - HH)];
    d[i] = (_Float16)v;
  }
}

__global__ void add3_kernel(const float* __restrict__ a, const float* __restrict__ b,
                            const float* __restrict__ c, float* __restrict__ o, size_t n) {
  size_t i = (size_t)blockIdx.x * blockDim.x + threadIdx.x;
  size_t st = (size_t)gridDim.x * blockDim.x;
  for (; i < n; i += st) o[i] = a[i] + b[i] + c[i];
}

__global__ void init_state_kernel(const float* __restrict__ h0l, const float* __restrict__ c0l,
                                  _Float16* __restrict__ h16, _Float16* __restrict__ c16,
                                  float* __restrict__ cS, float* __restrict__ hOutL,
                                  float* __restrict__ cOutL, size_t n) {
  size_t i = (size_t)blockIdx.x * blockDim.x + threadIdx.x;
  size_t st = (size_t)gridDim.x * blockDim.x;
  for (; i < n; i += st) {
    float h = h0l[i], c = c0l[i];
    h16[i] = (_Float16)h;
    c16[i] = (_Float16)c;
    cS[i] = c;
    hOutL[i] = h;
    cOutL[i] = c;
  }
}

__global__ void copy_out_kernel(const float* __restrict__ hOut, const float* __restrict__ cOut,
                                float* __restrict__ dst, size_t n) {
  size_t i = (size_t)blockIdx.x * blockDim.x + threadIdx.x;
  size_t st = (size_t)gridDim.x * blockDim.x;
  for (; i < n; i += st) {
    dst[i] = hOut[i];
    dst[n + i] = cOut[i];
  }
}

// ---------------- recurrent step kernel ----------------
//
// grid = (4 batch tiles, 32 H tiles, 2 directions), block = one wave32.
// Each wave computes a 16x16 (batch x H) state tile: the 4 gate tiles (i/f/g/o)
// via v_wmma_f32_16x16x32_f16 in three branch-free K segments
// (x·Wih, h·Whh, c·Wch), then the fused sigmoid/tanh cell update in fp32.

template <int KIN, bool F32OUT>
__global__ __launch_bounds__(32) void lstm_step(
    const _Float16* __restrict__ xin,   // [T*B, KIN] f16
    const _Float16* __restrict__ wih,   // [2, 4H, KIN] f16
    const _Float16* __restrict__ wrec,  // [2, 4H, 1024] f16 ([whh | wch] per gate row)
    const float* __restrict__ bias,     // [2, 4H] fused biases
    const _Float16* __restrict__ h16r,  // [2, B, H] read parity
    const _Float16* __restrict__ c16r,
    const float* __restrict__ cSr,
    _Float16* __restrict__ h16w,        // write parity
    _Float16* __restrict__ c16w,
    float* __restrict__ cSw,
    float* __restrict__ hOutL,          // [2, B, H] final-state staging
    float* __restrict__ cOutL,
    float* __restrict__ yf32,           // layer 1 output (d_out, f32)
    _Float16* __restrict__ y16,         // layer 0 output (y1, f16)
    int s) {
  const int lane = threadIdx.x;
  const int n = lane & 15;   // tile column / A-load row
  const int hi = lane >> 4;  // half-wave selector
  const int r = blockIdx.x;  // batch tile
  const int j = blockIdx.y;  // H tile
  const int d = blockIdx.z;  // direction
  const int t = d ? (TT - 1 - s) : s;
  const int am = r * 16 + n;        // A row (batch) loaded by this lane
  const int hcol = j * 16 + n;      // output column

  // --- issue independent scattered loads early (overlap with GEMM) ---
  const size_t sbase = (size_t)d * BB * HH + (size_t)(r * 16 + hi * 8) * HH + hcol;
  float cp[8];
#pragma unroll
  for (int v = 0; v < 8; ++v) cp[v] = cSr[sbase + (size_t)v * HH];
  const float* bb = bias + (size_t)d * FOURH;
  const float bi = bb[hcol];
  const float bfv = bb[HH + hcol];
  const float bg = bb[2 * HH + hcol];
  const float bo = bb[3 * HH + hcol];

  v8f acc0 = {}, acc1 = {}, acc2 = {}, acc3 = {};

  // --- segment 1: x_t @ Wih^T  (K = KIN) ---
  {
    const _Float16* ap = xin + ((size_t)t * BB + am) * KIN + hi * 8;
    const _Float16* w0 = wih + ((size_t)d * FOURH + hcol) * KIN + hi * 8;
    const _Float16* w1 = w0 + (size_t)HH * KIN;
    const _Float16* w2 = w1 + (size_t)HH * KIN;
    const _Float16* w3 = w2 + (size_t)HH * KIN;
#pragma unroll 4
    for (int kb = 0; kb < KIN / 32; ++kb) {
      const int o = kb * 32;
      v16h a = ldA(ap + o);
      acc0 = WMMA_F16(a, ldB(w0 + o), acc0);
      acc1 = WMMA_F16(a, ldB(w1 + o), acc1);
      acc2 = WMMA_F16(a, ldB(w2 + o), acc2);
      acc3 = WMMA_F16(a, ldB(w3 + o), acc3);
    }
  }

  // --- segment 2: h_prev @ Whh^T  (K = 512, wrec cols [0,512)) ---
  {
    const _Float16* ap = h16r + ((size_t)d * BB + am) * HH + hi * 8;
    const _Float16* w0 = wrec + ((size_t)d * FOURH + hcol) * 1024 + hi * 8;
    const _Float16* w1 = w0 + (size_t)HH * 1024;
    const _Float16* w2 = w1 + (size_t)HH * 1024;
    const _Float16* w3 = w2 + (size_t)HH * 1024;
#pragma unroll 4
    for (int kb = 0; kb < HH / 32; ++kb) {
      const int o = kb * 32;
      v16h a = ldA(ap + o);
      acc0 = WMMA_F16(a, ldB(w0 + o), acc0);
      acc1 = WMMA_F16(a, ldB(w1 + o), acc1);
      acc2 = WMMA_F16(a, ldB(w2 + o), acc2);
      acc3 = WMMA_F16(a, ldB(w3 + o), acc3);
    }
  }

  // --- segment 3: c_prev @ Wch^T  (K = 512, wrec cols [512,1024)) ---
  {
    const _Float16* ap = c16r + ((size_t)d * BB + am) * HH + hi * 8;
    const _Float16* w0 = wrec + ((size_t)d * FOURH + hcol) * 1024 + 512 + hi * 8;
    const _Float16* w1 = w0 + (size_t)HH * 1024;
    const _Float16* w2 = w1 + (size_t)HH * 1024;
    const _Float16* w3 = w2 + (size_t)HH * 1024;
#pragma unroll 4
    for (int kb = 0; kb < HH / 32; ++kb) {
      const int o = kb * 32;
      v16h a = ldA(ap + o);
      acc0 = WMMA_F16(a, ldB(w0 + o), acc0);
      acc1 = WMMA_F16(a, ldB(w1 + o), acc1);
      acc2 = WMMA_F16(a, ldB(w2 + o), acc2);
      acc3 = WMMA_F16(a, ldB(w3 + o), acc3);
    }
  }

  // --- fused gate activation + peephole cell update (fp32 c path) ---
#pragma unroll
  for (int v = 0; v < 8; ++v) {
    const int m = r * 16 + hi * 8 + v;  // C/D layout: lanes>=16 hold rows M+8
    const size_t sidx = sbase + (size_t)v * HH;
    const float gi = sigmoidf_(acc0[v] + bi);
    const float gf = sigmoidf_(acc1[v] + bfv);
    const float gg = tanhf(acc2[v] + bg);
    const float go = sigmoidf_(acc3[v] + bo);
    const float cn = fmaf(gf, cp[v], gi * gg);
    const float hn = go * tanhf(cn);
    cSw[sidx] = cn;
    c16w[sidx] = (_Float16)cn;
    h16w[sidx] = (_Float16)hn;
    hOutL[sidx] = hn;
    cOutL[sidx] = cn;
    const size_t yidx = ((size_t)t * BB + m) * (2 * HH) + (size_t)d * HH + hcol;
    if (F32OUT)
      yf32[yidx] = hn;
    else
      y16[yidx] = (_Float16)hn;
  }
}

// ---------------- host launcher ----------------

extern "C" void kernel_launch(void* const* d_in, const int* in_sizes, int n_in,
                              void* d_out, int out_size, void* d_ws, size_t ws_size,
                              hipStream_t stream) {
  (void)in_sizes; (void)n_in; (void)out_size; (void)ws_size;

  const float* x    = (const float*)d_in[0];
  const float* h0   = (const float*)d_in[1];
  const float* c0   = (const float*)d_in[2];
  const float* wih0 = (const float*)d_in[3];
  const float* whh0 = (const float*)d_in[4];
  const float* wch0 = (const float*)d_in[5];
  const float* bih0 = (const float*)d_in[6];
  const float* bhh0 = (const float*)d_in[7];
  const float* bch0 = (const float*)d_in[8];
  const float* wih1 = (const float*)d_in[9];
  const float* whh1 = (const float*)d_in[10];
  const float* wch1 = (const float*)d_in[11];
  const float* bih1 = (const float*)d_in[12];
  const float* bhh1 = (const float*)d_in[13];
  const float* bch1 = (const float*)d_in[14];

  char* ws = (char*)d_ws;
  size_t off = 0;
  auto alloc = [&](size_t bytes) -> void* {
    off = (off + 255) & ~(size_t)255;
    void* p = ws + off;
    off += bytes;
    return p;
  };

  const size_t SN = 2 * (size_t)BB * HH;  // [dir][B][H] elems per parity/layer slot

  _Float16* x16    = (_Float16*)alloc((size_t)TT * BB * II * 2);
  _Float16* wih0h  = (_Float16*)alloc((size_t)2 * FOURH * II * 2);
  _Float16* wrec0h = (_Float16*)alloc((size_t)2 * FOURH * 1024 * 2);
  _Float16* wih1h  = (_Float16*)alloc((size_t)2 * FOURH * 1024 * 2);
  _Float16* wrec1h = (_Float16*)alloc((size_t)2 * FOURH * 1024 * 2);
  _Float16* y1h    = (_Float16*)alloc((size_t)TT * BB * 1024 * 2);
  float* bias0 = (float*)alloc((size_t)2 * FOURH * 4);
  float* bias1 = (float*)alloc((size_t)2 * FOURH * 4);

  _Float16* h16a[2]; _Float16* c16a[2]; float* cSa[2];
  for (int L = 0; L < 2; ++L) {
    h16a[L] = (_Float16*)alloc(2 * SN * sizeof(_Float16));  // [parity][dir][B][H]
    c16a[L] = (_Float16*)alloc(2 * SN * sizeof(_Float16));
    cSa[L]  = (float*)alloc(2 * SN * sizeof(float));
  }
  float* hOut = (float*)alloc(2 * SN * sizeof(float));  // [layer][dir][B][H]
  float* cOut = (float*)alloc(2 * SN * sizeof(float));

  auto nb = [](size_t n) { unsigned b = (unsigned)((n + 255) / 256); return b > 8192u ? 8192u : b; };

  // prep: f16 conversions, fused weights [whh|wch], fused bias, state init
  {
    size_t n = (size_t)TT * BB * II;
    cvt_f16_kernel<<<nb(n), 256, 0, stream>>>(x, x16, n);
    n = (size_t)2 * FOURH * II;
    cvt_f16_kernel<<<nb(n), 256, 0, stream>>>(wih0, wih0h, n);
    n = (size_t)2 * FOURH * 1024;
    build_wrec_kernel<<<nb(n), 256, 0, stream>>>(whh0, wch0, wrec0h, n);
    cvt_f16_kernel<<<nb(n), 256, 0, stream>>>(wih1, wih1h, n);
    build_wrec_kernel<<<nb(n), 256, 0, stream>>>(whh1, wch1, wrec1h, n);
    n = (size_t)2 * FOURH;
    add3_kernel<<<nb(n), 256, 0, stream>>>(bih0, bhh0, bch0, bias0, n);
    add3_kernel<<<nb(n), 256, 0, stream>>>(bih1, bhh1, bch1, bias1, n);
    for (int L = 0; L < 2; ++L)
      init_state_kernel<<<nb(SN), 256, 0, stream>>>(h0 + L * SN, c0 + L * SN,
                                                    h16a[L], c16a[L], cSa[L],
                                                    hOut + L * SN, cOut + L * SN, SN);
  }

  const dim3 sg(4, 32, 2), sb(32, 1, 1);

  // layer 0: x (KIN=512) -> y1 (f16)
  for (int s = 0; s < TT; ++s) {
    const int pr = s & 1, pw = 1 - pr;
    lstm_step<II, false><<<sg, sb, 0, stream>>>(
        x16, wih0h, wrec0h, bias0,
        h16a[0] + pr * SN, c16a[0] + pr * SN, cSa[0] + pr * SN,
        h16a[0] + pw * SN, c16a[0] + pw * SN, cSa[0] + pw * SN,
        hOut, cOut, nullptr, y1h, s);
  }
  // layer 1: y1 (KIN=1024) -> y2 (f32, straight into d_out)
  for (int s = 0; s < TT; ++s) {
    const int pr = s & 1, pw = 1 - pr;
    lstm_step<1024, true><<<sg, sb, 0, stream>>>(
        y1h, wih1h, wrec1h, bias1,
        h16a[1] + pr * SN, c16a[1] + pr * SN, cSa[1] + pr * SN,
        h16a[1] + pw * SN, c16a[1] + pw * SN, cSa[1] + pw * SN,
        hOut + SN, cOut + SN, (float*)d_out, nullptr, s);
  }

  // final h/c: [2 layers][2 dirs][B][H] each, appended after y2
  float* tail = (float*)d_out + (size_t)TT * BB * 2 * HH;
  copy_out_kernel<<<nb(2 * SN), 256, 0, stream>>>(hOut, cOut, tail, 2 * SN);
}
